// DLGConv_78769700208782
// MI455X (gfx1250) — compile-verified, matrix operand order checked by reading, not run on hardware
//
#include <hip/hip_runtime.h>
#include <hip/hip_bf16.h>

typedef __attribute__((ext_vector_type(16))) _Float16 v16h;
typedef __attribute__((ext_vector_type(8)))  float    v8f;

#define K_HEADS 4
#define D_FEAT  16
#define NODE_STRIDE (K_HEADS * D_FEAT)   // 64 floats per node
#define NEG_SLOPE 0.01f
#define EPS 1e-12f

// ---- order-preserving float<->uint encoding so atomicMax(uint) == float max ----
__device__ __forceinline__ unsigned enc_f32(float f) {
    unsigned u = __float_as_uint(f);
    return (u & 0x80000000u) ? ~u : (u | 0x80000000u);
}
__device__ __forceinline__ float dec_f32(unsigned u) {
    return (u & 0x80000000u) ? __uint_as_float(u ^ 0x80000000u)
                             : __uint_as_float(~u);
}

// ---- pass 0: zero the m-accumulator (d_out), denom, and encoded smax ----
__global__ void k_init(float* __restrict__ out, unsigned* __restrict__ smax,
                       float* __restrict__ denom, int n_out, int n_nk) {
    int i = blockIdx.x * blockDim.x + threadIdx.x;
    if (i < n_out) out[i] = 0.0f;
    if (i < n_nk) { smax[i] = 0u; denom[i] = 0.0f; }   // enc 0 == lowest == -inf
}

// ---- pass 1: WMMA edge scores (16 edges per wave), leaky-ReLU, atomic seg-max ----
// score[e,k] = sum_j f_src[e,k,j]*W[k,j] + f_dst[e,k,j]*W[k,D+j]
// Folded into D = A(16x32) x B(32x16) accumulated over 4 head-chunks:
// chunk k: A = [f_src[:,k,0:16] | f_dst[:,k,0:16]], B col n = (n==k) ? W[k,:] : 0
__global__ void k_score(const float* __restrict__ f_in,
                        const int*   __restrict__ edges,
                        const float* __restrict__ attn_w,
                        unsigned*    __restrict__ smax_enc,
                        float*       __restrict__ scores,
                        int E) {
    const int wave = threadIdx.x >> 5;
    const int lane = threadIdx.x & 31;
    const int tile = blockIdx.x * (blockDim.x >> 5) + wave;
    const int nTiles = (E + 15) >> 4;
    if (tile >= nTiles) return;                 // wave-uniform: EXEC all-1 for WMMA

    const int e0   = tile << 4;
    const int col  = lane & 15;                 // row (A) / column (B,D) index
    const int half = lane >> 4;                 // which K-slab this lane holds
    const int off  = half << 3;                 // 0 or 8

    int e = e0 + col; if (e >= E) e = E - 1;    // clamped load, stores guarded below
    const int s = edges[2 * e];
    const int d = edges[2 * e + 1];

    v8f c = {};
    #pragma unroll
    for (int k = 0; k < K_HEADS; ++k) {
        const float* ps = f_in + ((size_t)s * NODE_STRIDE + k * D_FEAT + off);
        const float* pd = f_in + ((size_t)d * NODE_STRIDE + k * D_FEAT + off);
        v16h a;
        #pragma unroll
        for (int i = 0; i < 8; ++i) {
            a[i]     = (_Float16)ps[i];         // A[:, 0:16]  = f_src slab
            a[8 + i] = (_Float16)pd[i];         // A[:, 16:32] = f_dst slab
        }
        const float msk = (col == k) ? 1.0f : 0.0f;   // data select only, no EXEC change
        v16h b;
        #pragma unroll
        for (int i = 0; i < 8; ++i) {
            b[i]     = (_Float16)(attn_w[k * 32 + off + i] * msk);
            b[8 + i] = (_Float16)(attn_w[k * 32 + 16 + off + i] * msk);
        }
        c = __builtin_amdgcn_wmma_f32_16x16x32_f16(false, a, false, b,
                                                   (short)0, c, false, false);
    }

    // D layout: VGPR r -> M = r + 8*half, N = col. Columns 0..3 hold the 4 heads.
    if (col < K_HEADS) {
        #pragma unroll
        for (int r = 0; r < 8; ++r) {
            const int e2 = e0 + half * 8 + r;
            if (e2 < E) {
                float sc = c[r];
                sc = sc >= 0.0f ? sc : NEG_SLOPE * sc;        // leaky ReLU pre-max
                scores[(size_t)e2 * K_HEADS + col] = sc;
                const int s2 = edges[2 * e2];
                atomicMax(&smax_enc[s2 * K_HEADS + col], enc_f32(sc));
            }
        }
    }
}

// ---- pass 2: ex = exp(score - smax[src]); denom[src] += ex (atomic) ----
__global__ void k_exp(const int* __restrict__ edges,
                      const unsigned* __restrict__ smax_enc,
                      float* __restrict__ scores,
                      float* __restrict__ denom, int EK) {
    int i = blockIdx.x * blockDim.x + threadIdx.x;
    if (i >= EK) return;
    const int e = i >> 2, k = i & 3;
    const int s = edges[2 * e];
    const float mx = dec_f32(smax_enc[s * K_HEADS + k]);
    const float ex = __expf(scores[i] - mx);
    scores[i] = ex;
    atomicAdd(&denom[s * K_HEADS + k], ex);
}

// ---- pass 3: m[src,k,d] += f_dst[k,d] * ex / denom[src,k] (atomic scatter) ----
__global__ void k_aggregate(const float* __restrict__ f_in,
                            const int*   __restrict__ edges,
                            const float* __restrict__ scores,
                            const float* __restrict__ denom,
                            float* __restrict__ out, int ED) {
    int i = blockIdx.x * blockDim.x + threadIdx.x;
    if (i >= ED) return;
    const int e = i >> 4, d = i & 15;
    const int s  = edges[2 * e];
    const int dt = edges[2 * e + 1];
    #pragma unroll
    for (int k = 0; k < K_HEADS; ++k) {
        const float ex  = scores[(size_t)e * K_HEADS + k];
        const float den = denom[s * K_HEADS + k];
        const float f   = f_in[(size_t)dt * NODE_STRIDE + k * D_FEAT + d];
        atomicAdd(&out[(size_t)s * NODE_STRIDE + k * D_FEAT + d], f * (ex / den));
    }
}

// ---- pass 4: in-place L2 normalize over head axis: out /= max(||m||_k, eps) ----
__global__ void k_l2norm(float* __restrict__ out, int ND) {
    int i = blockIdx.x * blockDim.x + threadIdx.x;
    if (i >= ND) return;
    const int n = i >> 4, d = i & 15;
    float v[K_HEADS], ss = 0.0f;
    #pragma unroll
    for (int k = 0; k < K_HEADS; ++k) {
        v[k] = out[(size_t)n * NODE_STRIDE + k * D_FEAT + d];
        ss += v[k] * v[k];
    }
    const float inv = 1.0f / fmaxf(sqrtf(ss), EPS);
    #pragma unroll
    for (int k = 0; k < K_HEADS; ++k)
        out[(size_t)n * NODE_STRIDE + k * D_FEAT + d] = v[k] * inv;
}

extern "C" void kernel_launch(void* const* d_in, const int* in_sizes, int n_in,
                              void* d_out, int out_size, void* d_ws, size_t ws_size,
                              hipStream_t stream) {
    const float* f_in   = (const float*)d_in[0];
    const int*   edges  = (const int*)d_in[1];
    const float* attn_w = (const float*)d_in[2];
    float* out = (float*)d_out;

    const int N = in_sizes[0] / NODE_STRIDE;   // nodes
    const int E = in_sizes[1] / 2;             // edges
    const int NK  = N * K_HEADS;
    const int NKD = N * NODE_STRIDE;
    const int EK  = E * K_HEADS;
    const int ED  = E * D_FEAT;

    // workspace layout: [smax_enc: NK u32][denom: NK f32][scores: EK f32]
    unsigned* smax_enc = (unsigned*)d_ws;
    float*    denom    = (float*)d_ws + NK;
    float*    scores   = (float*)d_ws + 2 * (size_t)NK;

    const int T = 256;
    k_init<<<(NKD + T - 1) / T, T, 0, stream>>>(out, smax_enc, denom, NKD, NK);

    const int nTiles = (E + 15) / 16;
    const int wavesPerBlock = T / 32;
    k_score<<<(nTiles + wavesPerBlock - 1) / wavesPerBlock, T, 0, stream>>>(
        f_in, edges, attn_w, smax_enc, scores, E);

    k_exp<<<(EK + T - 1) / T, T, 0, stream>>>(edges, smax_enc, scores, denom, EK);

    k_aggregate<<<(ED + T - 1) / T, T, 0, stream>>>(f_in, edges, scores, denom, out, ED);

    k_l2norm<<<(N * D_FEAT + T - 1) / T, T, 0, stream>>>(out, N * D_FEAT);
}
